// TCRInformer_39101382263286
// MI455X (gfx1250) — compile-verified, matrix operand order checked by reading, not run on hardware
//
#include <hip/hip_runtime.h>
#include <math.h>

typedef __attribute__((ext_vector_type(16))) __bf16 bf16x16;
typedef __attribute__((ext_vector_type(8)))  __bf16 bf16x8;
typedef __attribute__((ext_vector_type(8)))  float  f32x8;

#define NB      8      // batch
#define DEC_LEN 256    // LABEL_LEN + PRED_LEN
#define PRED    128
#define DM      512    // d_model
#define DFF     2048   // d_ff
#define NH      8      // heads
#define HD      64     // head dim
#define COUT    7
#define SPK     30     // sample_k = u = min(5*ceil(ln 256), 256) = 30

// ---------------------------------------------------------------- helpers
__device__ __forceinline__ __bf16 f2bf(float f) {
  unsigned u = __builtin_bit_cast(unsigned, f);
  unsigned r = (u + 0x7fffu + ((u >> 16) & 1u)) >> 16;   // RNE
  unsigned short h = (unsigned short)r;
  return __builtin_bit_cast(__bf16, h);
}

// ---------------------------------------------------------------- f32 -> bf16
__global__ void k_cvt_bf16(const float* __restrict__ in, __bf16* __restrict__ out, int n) {
  int i = blockIdx.x * blockDim.x + threadIdx.x;
  if (i < n) out[i] = f2bf(in[i]);
}

// ---------------------------------------------------------------- WMMA GEMM
// C[M,N] = A[M,K] (f32 -> bf16 in staging) * Bw[K,N] (bf16) + bias[N]
// (+ res[M,N]) (+ exact GELU).  M%64==0, N%128==0, K%32==0.
// Double-buffered LDS: one barrier per K-step; global loads for tile i+1
// issue before the WMMAs of tile i; prefetch pulls tile i+2.
template <bool HAS_RES, bool HAS_GELU>
__global__ __launch_bounds__(256)
void k_gemm_wmma(const float* __restrict__ A, const __bf16* __restrict__ Bw,
                 const float* __restrict__ bias, const float* __restrict__ res,
                 float* __restrict__ C, int M, int N, int K) {
  __shared__ __align__(16) __bf16 As[2][64 * 40];    // [m][k], stride 40
  __shared__ __align__(16) __bf16 Bs[2][128 * 40];   // [n][k], stride 40 (transposed)

  const int tid  = threadIdx.x;
  const int lane = tid & 31;
  const int wv   = tid >> 5;          // 8 waves
  const int wm   = wv & 1;            // 2 wave-rows (32 rows each)
  const int wn   = wv >> 1;           // 4 wave-cols (32 cols each)
  const int tileM = blockIdx.y * 64;
  const int tileN = blockIdx.x * 128;

  f32x8 acc[2][2];
#pragma unroll
  for (int i = 0; i < 2; i++)
#pragma unroll
    for (int j = 0; j < 2; j++)
#pragma unroll
      for (int v = 0; v < 8; v++) acc[i][j][v] = 0.0f;

  const int ar = tid >> 2;            // 0..63 : A row
  const int ac = (tid & 3) * 8;       // 0,8,16,24 : A col group
  const int br = tid >> 3;            // 0..31 : B k-row
  const int bc = (tid & 7) * 16;      // 0..112 : B n-col group

  const int row = lane & 15;
  const int kh  = (lane >> 4) * 8;    // K-half select per ISA 16-bit A/B layout

  const float*  aptr = A  + (size_t)(tileM + ar) * K + ac;
  const __bf16* bptr = Bw + (size_t)br * N + tileN + bc;
  const size_t  bstep = (size_t)32 * N;          // k-step for B rows

  // ---- prologue: stage tile 0 into buffer 0
  {
    const float4* s4 = (const float4*)aptr;
    float4 f0 = s4[0], f1 = s4[1];
    bf16x8 v;
    v[0] = f2bf(f0.x); v[1] = f2bf(f0.y); v[2] = f2bf(f0.z); v[3] = f2bf(f0.w);
    v[4] = f2bf(f1.x); v[5] = f2bf(f1.y); v[6] = f2bf(f1.z); v[7] = f2bf(f1.w);
    *(bf16x8*)&As[0][ar * 40 + ac] = v;
    const uint4* t4 = (const uint4*)bptr;
    uint4 p0 = t4[0], p1 = t4[1];
    const __bf16* e0 = (const __bf16*)&p0;
    const __bf16* e1 = (const __bf16*)&p1;
#pragma unroll
    for (int i = 0; i < 8; i++) Bs[0][(bc + i) * 40 + br] = e0[i];
#pragma unroll
    for (int i = 0; i < 8; i++) Bs[0][(bc + 8 + i) * 40 + br] = e1[i];
  }

  const int nk = K >> 5;
  for (int it = 0; it < nk; ++it) {
    __syncthreads();                   // LDS[buf] ready; prior reads of LDS[buf^1] done
    const int buf = it & 1;
    const bool more = (it + 1 < nk);

    // prefetch tile i+2 toward WGP (global_prefetch_b8)
    if (it + 2 < nk) {
      __builtin_prefetch(aptr + (size_t)(it + 2) * 32, 0, 0);
      __builtin_prefetch(bptr + (size_t)(it + 2) * bstep, 0, 0);
    }

    // issue global loads for tile i+1 before computing tile i
    float4 f0 = {}, f1 = {};
    uint4  p0 = {}, p1 = {};
    if (more) {
      const float4* s4 = (const float4*)(aptr + (size_t)(it + 1) * 32);
      f0 = s4[0]; f1 = s4[1];
      const uint4* t4 = (const uint4*)(bptr + (size_t)(it + 1) * bstep);
      p0 = t4[0]; p1 = t4[1];
    }

    // compute tile i from LDS[buf]
    bf16x16 afrag[2], bfrag[2];
#pragma unroll
    for (int i = 0; i < 2; i++) {
      const __bf16* p = &As[buf][(wm * 32 + i * 16 + row) * 40];
      bf16x8 lo = *(const bf16x8*)&p[kh];
      bf16x8 hi = *(const bf16x8*)&p[16 + kh];
      afrag[i] = __builtin_shufflevector(lo, hi, 0,1,2,3,4,5,6,7,8,9,10,11,12,13,14,15);
    }
#pragma unroll
    for (int j = 0; j < 2; j++) {
      const __bf16* p = &Bs[buf][(wn * 32 + j * 16 + row) * 40];
      bf16x8 lo = *(const bf16x8*)&p[kh];
      bf16x8 hi = *(const bf16x8*)&p[16 + kh];
      bfrag[j] = __builtin_shufflevector(lo, hi, 0,1,2,3,4,5,6,7,8,9,10,11,12,13,14,15);
    }
#pragma unroll
    for (int i = 0; i < 2; i++)
#pragma unroll
      for (int j = 0; j < 2; j++)
        acc[i][j] = __builtin_amdgcn_wmma_f32_16x16x32_bf16(
            false, afrag[i], false, bfrag[j], (short)0, acc[i][j], false, false);

    // stage tile i+1 into LDS[buf^1]
    if (more) {
      bf16x8 v;
      v[0] = f2bf(f0.x); v[1] = f2bf(f0.y); v[2] = f2bf(f0.z); v[3] = f2bf(f0.w);
      v[4] = f2bf(f1.x); v[5] = f2bf(f1.y); v[6] = f2bf(f1.z); v[7] = f2bf(f1.w);
      *(bf16x8*)&As[buf ^ 1][ar * 40 + ac] = v;
      const __bf16* e0 = (const __bf16*)&p0;
      const __bf16* e1 = (const __bf16*)&p1;
#pragma unroll
      for (int i = 0; i < 8; i++) Bs[buf ^ 1][(bc + i) * 40 + br] = e0[i];
#pragma unroll
      for (int i = 0; i < 8; i++) Bs[buf ^ 1][(bc + 8 + i) * 40 + br] = e1[i];
    }
  }

  // epilogue: C/D layout — VGPR r: lanes 0-15 M=r, lanes 16-31 M=8+r; N=lane&15
  const int mlo = (lane >> 4) * 8;
  const int nn  = lane & 15;
#pragma unroll
  for (int i = 0; i < 2; i++)
#pragma unroll
    for (int j = 0; j < 2; j++) {
      const int mb = tileM + wm * 32 + i * 16 + mlo;
      const int n  = tileN + wn * 32 + j * 16 + nn;
      const float bn = bias[n];
      float r[8];
#pragma unroll
      for (int v = 0; v < 8; v++) {
        r[v] = acc[i][j][v] + bn;
        if (HAS_RES) r[v] += res[(size_t)(mb + v) * N + n];
      }
#pragma unroll
      for (int v = 0; v < 8; v++) {
        float x = r[v];
        if (HAS_GELU) x = 0.5f * x * (1.0f + erff(x * 0.70710678118f));
        C[(size_t)(mb + v) * N + n] = x;
      }
    }
}

// ---------------------------------------------------------------- embedding
__global__ void k_embed(const float* __restrict__ xin, const float* __restrict__ w,
                        const float* __restrict__ b, const float* __restrict__ mn,
                        const float* __restrict__ hr, const float* __restrict__ dy,
                        const float* __restrict__ mo, float* __restrict__ out,
                        int L, int cin) {
  int id = blockIdx.x * blockDim.x + threadIdx.x;
  int total = NB * L * DM;
  if (id >= total) return;
  int d = id % DM;
  int l = (id / DM) % L;
  int bb = id / (DM * L);
  float s = b[d];
  const float* xr = xin + (size_t)(bb * L + l) * cin;
  for (int c = 0; c < cin; c++) s += xr[c] * w[c * DM + d];
  int seg = d >> 7, dd = d & 127;
  float t;
  if      (seg == 0) t = mn[(l % 60) * 128 + dd];
  else if (seg == 1) t = hr[((l / 60) % 24) * 128 + dd];
  else if (seg == 2) t = dy[((l / 1440) % 32) * 128 + dd];
  else               t = mo[((l / 46080) % 13) * 128 + dd];
  out[id] = s + t;
}

// ---------------------------------------------------------------- layernorm
__global__ __launch_bounds__(128)
void k_layernorm(const float* __restrict__ x, const float* __restrict__ g,
                 const float* __restrict__ be, float* __restrict__ out) {
  __shared__ float red[128];
  const int tid = threadIdx.x;
  const float* xr = x + (size_t)blockIdx.x * DM;
  float v[4];
  float s = 0.0f;
#pragma unroll
  for (int i = 0; i < 4; i++) { v[i] = xr[tid + i * 128]; s += v[i]; }
  red[tid] = s; __syncthreads();
  for (int off = 64; off > 0; off >>= 1) {
    if (tid < off) red[tid] += red[tid + off];
    __syncthreads();
  }
  float mean = red[0] * (1.0f / DM); __syncthreads();
  float s2 = 0.0f;
#pragma unroll
  for (int i = 0; i < 4; i++) { float d = v[i] - mean; s2 += d * d; }
  red[tid] = s2; __syncthreads();
  for (int off = 64; off > 0; off >>= 1) {
    if (tid < off) red[tid] += red[tid + off];
    __syncthreads();
  }
  float inv = rsqrtf(red[0] * (1.0f / DM) + 1e-5f);
  float* o = out + (size_t)blockIdx.x * DM;
#pragma unroll
  for (int i = 0; i < 4; i++) {
    int d = tid + i * 128;
    o[d] = (v[i] - mean) * inv * g[d] + be[d];
  }
}

// ---------------------------------------------------------------- sparsity metric
__global__ void k_metric(const float* __restrict__ Q, const float* __restrict__ Kk,
                         float* __restrict__ Mm, int L, int sk) {
  int id = blockIdx.x * blockDim.x + threadIdx.x;
  int total = NB * NH * L;
  if (id >= total) return;
  int l = id % L;
  int h = (id / L) % NH;
  int b = id / (L * NH);
  const float* q = Q + (size_t)(b * L + l) * DM + h * HD;
  float mx = -1e30f, sm = 0.0f;
  for (int s = 0; s < sk; s++) {
    unsigned ks = ((unsigned)l * 2654435761u + (unsigned)s * 40503u + 12345u) % (unsigned)L;
    const float* kr = Kk + (size_t)(b * L + ks) * DM + h * HD;
    float d = 0.0f;
    for (int e = 0; e < HD; e++) d += q[e] * kr[e];
    mx = fmaxf(mx, d); sm += d;
  }
  Mm[id] = mx - sm / (float)L;
}

// ---------------------------------------------------------------- top-u (wave32, register-resident)
__global__ __launch_bounds__(32)
void k_topk(const float* __restrict__ Mm, int* __restrict__ TI, int L, int u) {
  const int bh = blockIdx.x;
  const int lane = threadIdx.x;
  const float* base = Mm + (size_t)bh * L;
  const int n = L >> 5;                 // values per lane (256/32 = 8)
  float mv[8];
#pragma unroll
  for (int i = 0; i < 8; i++) mv[i] = (i < n) ? base[lane + (i << 5)] : -1e30f;
  for (int t = 0; t < u; t++) {
    float best = -1e30f; int bi = 0x7fffffff;
#pragma unroll
    for (int i = 0; i < 8; i++) {
      int idx = lane + (i << 5);
      if (mv[i] > best || (mv[i] == best && idx < bi)) { best = mv[i]; bi = idx; }
    }
#pragma unroll
    for (int off = 16; off > 0; off >>= 1) {
      float ov = __shfl_xor(best, off);
      int   oi = __shfl_xor(bi, off);
      if (ov > best || (ov == best && oi < bi)) { best = ov; bi = oi; }
    }
    if (lane == 0) TI[(size_t)bh * u + t] = bi;
    if ((bi & 31) == lane) {
      int slot = bi >> 5;
#pragma unroll
      for (int i = 0; i < 8; i++) if (i == slot) mv[i] = -1e30f;
    }
  }
}

// ---------------------------------------------------------------- ctx = cumsum(V, axis=L)
__global__ void k_ctx(const float* __restrict__ V, float* __restrict__ out, int L) {
  int id = blockIdx.x * blockDim.x + threadIdx.x;
  int total = NB * NH * HD;
  if (id >= total) return;
  int e = id % HD;
  int h = (id / HD) % NH;
  int b = id / (HD * NH);
  const float* v = V + (size_t)b * L * DM + h * HD + e;
  float* o = out + (size_t)b * L * DM + h * HD + e;
  float run = 0.0f;
  for (int l = 0; l < L; l++) {
    run += v[(size_t)l * DM];
    o[(size_t)l * DM] = run;
  }
}

// ---------------------------------------------------------------- softmax over full keys for selected queries
__global__ __launch_bounds__(256)
void k_sparse_attn(const float* __restrict__ Q, const float* __restrict__ Kk,
                   const float* __restrict__ V, const int* __restrict__ TI,
                   float* __restrict__ out, int L, int u) {
  __shared__ float qs[HD];
  __shared__ float red[256];
  __shared__ float sc[1024];            // L <= 1024
  const int tid = threadIdx.x;
  const int ui = blockIdx.x, h = blockIdx.y, b = blockIdx.z;
  const int bh = b * NH + h;
  const int ti = TI[(size_t)bh * u + ui];
  if (tid < HD) qs[tid] = Q[(size_t)(b * L + ti) * DM + h * HD + tid];
  __syncthreads();

  const float scale = 0.125f;           // 1/sqrt(64)
  float lmax = -1e30f;
  for (int k = tid; k < L; k += 256) {
    const float* kr = Kk + (size_t)(b * L + k) * DM + h * HD;
    float d = 0.0f;
    for (int e = 0; e < HD; e++) d += qs[e] * kr[e];
    d *= scale;
    sc[k] = d;
    lmax = fmaxf(lmax, d);
  }
  red[tid] = lmax; __syncthreads();
  for (int off = 128; off > 0; off >>= 1) {
    if (tid < off) red[tid] = fmaxf(red[tid], red[tid + off]);
    __syncthreads();
  }
  float mx = red[0]; __syncthreads();
  float lsum = 0.0f;
  for (int k = tid; k < L; k += 256) {
    float e = __expf(sc[k] - mx);
    sc[k] = e;
    lsum += e;
  }
  red[tid] = lsum; __syncthreads();
  for (int off = 128; off > 0; off >>= 1) {
    if (tid < off) red[tid] += red[tid + off];
    __syncthreads();
  }
  float denom = red[0];
  if (tid < HD) {
    const float* vb = V + (size_t)b * L * DM + h * HD + tid;
    float a = 0.0f;
    for (int k = 0; k < L; k++) a += sc[k] * vb[(size_t)k * DM];
    out[(size_t)(b * L + ti) * DM + h * HD + tid] = a / denom;
  }
}

// ---------------------------------------------------------------- projection + slice [:, -128:, :]
__global__ void k_proj(const float* __restrict__ y, const float* __restrict__ w,
                       const float* __restrict__ bias, float* __restrict__ out) {
  int id = blockIdx.x * blockDim.x + threadIdx.x;
  if (id >= NB * PRED * COUT) return;
  int c = id % COUT;
  int i = (id / COUT) % PRED;
  int b = id / (COUT * PRED);
  int l = (DEC_LEN - PRED) + i;
  const float* yr = y + (size_t)(b * DEC_LEN + l) * DM;
  float s = bias[c];
  for (int d = 0; d < DM; d++) s += yr[d] * w[d * COUT + c];
  out[id] = s;
}

// ================================================================ host
static inline char* bump(char*& p, size_t bytes) {
  char* r = p;
  p += (bytes + 255) & ~(size_t)255;
  return r;
}

static void launch_gemm(const float* A, const __bf16* Bw, const float* bias,
                        const float* res, float* C, int M, int N, int K,
                        int gelu, hipStream_t s) {
  dim3 g(N / 128, M / 64), b(256);
  if (res && gelu)
    hipLaunchKernelGGL((k_gemm_wmma<true, true>),   g, b, 0, s, A, Bw, bias, res, C, M, N, K);
  else if (res)
    hipLaunchKernelGGL((k_gemm_wmma<true, false>),  g, b, 0, s, A, Bw, bias, res, C, M, N, K);
  else if (gelu)
    hipLaunchKernelGGL((k_gemm_wmma<false, true>),  g, b, 0, s, A, Bw, bias, res, C, M, N, K);
  else
    hipLaunchKernelGGL((k_gemm_wmma<false, false>), g, b, 0, s, A, Bw, bias, res, C, M, N, K);
}

extern "C" void kernel_launch(void* const* d_in, const int* in_sizes, int n_in,
                              void* d_out, int out_size, void* d_ws, size_t ws_size,
                              hipStream_t stream) {
  (void)in_sizes; (void)n_in; (void)out_size; (void)ws_size;
  // setup_inputs order: x_enc, x_dec, enc_w, enc_b, dec_w, dec_b, min, hour, day,
  // month, proj_w, proj_b, enc_layer0[16], enc_layer1[16], dec_layer0[16].
  // KEY OPTIMIZATION: the encoder output is dead in the reference (_forward never
  // feeds x into the decoder), so we run only: dec embed -> dec layer -> proj.
  const float* x_dec  = (const float*)d_in[1];
  const float* dec_w  = (const float*)d_in[4];
  const float* dec_b  = (const float*)d_in[5];
  const float* mn_emb = (const float*)d_in[6];
  const float* hr_emb = (const float*)d_in[7];
  const float* dy_emb = (const float*)d_in[8];
  const float* mo_emb = (const float*)d_in[9];
  const float* proj_w = (const float*)d_in[10];
  const float* proj_b = (const float*)d_in[11];
  const int P = 44;  // dec_layers[0] base
  const float* wq = (const float*)d_in[P+0];  const float* bq  = (const float*)d_in[P+1];
  const float* wk = (const float*)d_in[P+2];  const float* bk  = (const float*)d_in[P+3];
  const float* wvp= (const float*)d_in[P+4];  const float* bv  = (const float*)d_in[P+5];
  const float* wo = (const float*)d_in[P+6];  const float* bo  = (const float*)d_in[P+7];
  const float* w1 = (const float*)d_in[P+8];  const float* b1f = (const float*)d_in[P+9];
  const float* w2 = (const float*)d_in[P+10]; const float* b2f = (const float*)d_in[P+11];
  const float* g1 = (const float*)d_in[P+12]; const float* be1 = (const float*)d_in[P+13];
  const float* g2 = (const float*)d_in[P+14]; const float* be2 = (const float*)d_in[P+15];

  const int L = DEC_LEN;          // 256
  const int M = NB * L;           // 2048
  const int U = SPK;              // u = sample_k = 30

  char* p = (char*)d_ws;
  const size_t AELEM = (size_t)M * DM;          // 1M floats
  float*  XA  = (float*)bump(p, AELEM * 4);     // activation
  float*  X1  = (float*)bump(p, AELEM * 4);     // post-LN1
  float*  TMP = (float*)bump(p, AELEM * 4);
  float*  QF  = (float*)bump(p, AELEM * 4);
  float*  KF  = (float*)bump(p, AELEM * 4);
  float*  VF  = (float*)bump(p, AELEM * 4);
  float*  AT  = (float*)bump(p, AELEM * 4);
  float*  FF  = (float*)bump(p, (size_t)M * DFF * 4);
  float*  MM  = (float*)bump(p, (size_t)NB * NH * L * 4);
  int*    TI  = (int*)  bump(p, (size_t)NB * NH * U * 4);
  __bf16* wq_b = (__bf16*)bump(p, (size_t)DM * DM * 2);
  __bf16* wk_b = (__bf16*)bump(p, (size_t)DM * DM * 2);
  __bf16* wv_b = (__bf16*)bump(p, (size_t)DM * DM * 2);
  __bf16* wo_b = (__bf16*)bump(p, (size_t)DM * DM * 2);
  __bf16* w1_b = (__bf16*)bump(p, (size_t)DM * DFF * 2);
  __bf16* w2_b = (__bf16*)bump(p, (size_t)DFF * DM * 2);

  const int T = 256;
  // weight conversion (f32 -> bf16), every call (deterministic)
  hipLaunchKernelGGL(k_cvt_bf16, dim3((DM*DM + T-1)/T), dim3(T), 0, stream, wq,  wq_b, DM*DM);
  hipLaunchKernelGGL(k_cvt_bf16, dim3((DM*DM + T-1)/T), dim3(T), 0, stream, wk,  wk_b, DM*DM);
  hipLaunchKernelGGL(k_cvt_bf16, dim3((DM*DM + T-1)/T), dim3(T), 0, stream, wvp, wv_b, DM*DM);
  hipLaunchKernelGGL(k_cvt_bf16, dim3((DM*DM + T-1)/T), dim3(T), 0, stream, wo,  wo_b, DM*DM);
  hipLaunchKernelGGL(k_cvt_bf16, dim3((DM*DFF + T-1)/T), dim3(T), 0, stream, w1, w1_b, DM*DFF);
  hipLaunchKernelGGL(k_cvt_bf16, dim3((DFF*DM + T-1)/T), dim3(T), 0, stream, w2, w2_b, DFF*DM);

  // decoder embedding
  hipLaunchKernelGGL(k_embed, dim3((NB*L*DM + T-1)/T), dim3(T), 0, stream,
                     x_dec, dec_w, dec_b, mn_emb, hr_emb, dy_emb, mo_emb, XA, L, 7);

  // ---- decoder layer (self-attention, _enc_layer semantics) ----
  launch_gemm(XA, wq_b, bq, nullptr, QF, M, DM, DM, 0, stream);
  launch_gemm(XA, wk_b, bk, nullptr, KF, M, DM, DM, 0, stream);
  launch_gemm(XA, wv_b, bv, nullptr, VF, M, DM, DM, 0, stream);

  hipLaunchKernelGGL(k_metric, dim3((NB*NH*L + T-1)/T), dim3(T), 0, stream, QF, KF, MM, L, SPK);
  hipLaunchKernelGGL(k_topk, dim3(NB*NH), dim3(32), 0, stream, MM, TI, L, U);
  hipLaunchKernelGGL(k_ctx, dim3((NB*NH*HD + 127)/128), dim3(128), 0, stream, VF, AT, L);
  hipLaunchKernelGGL(k_sparse_attn, dim3(U, NH, NB), dim3(256), 0, stream, QF, KF, VF, TI, AT, L, U);

  launch_gemm(AT, wo_b, bo, XA, TMP, M, DM, DM, 0, stream);              // + residual x
  hipLaunchKernelGGL(k_layernorm, dim3(M), dim3(128), 0, stream, TMP, g1, be1, X1);

  launch_gemm(X1, w1_b, b1f, nullptr, FF, M, DFF, DM, 1, stream);        // GELU fused
  launch_gemm(FF, w2_b, b2f, X1, TMP, M, DM, DFF, 0, stream);            // + residual x1
  hipLaunchKernelGGL(k_layernorm, dim3(M), dim3(128), 0, stream, TMP, g2, be2, XA);

  // projection + slice
  hipLaunchKernelGGL(k_proj, dim3((NB*PRED*COUT + T-1)/T), dim3(T), 0, stream,
                     XA, proj_w, proj_b, (float*)d_out);
}
